// FLASH_ShareA_FFConvM_46797963657223
// MI455X (gfx1250) — matrix operand order repaired and checked by reading, not compiled, not run on hardware
//
#include <hip/hip_runtime.h>
#include <hip/hip_bf16.h>
#include <math.h>
#include <stdint.h>

// ---------------------------------------------------------------------------
// FLASH ShareA-FFConvM block for MI455X (gfx1250, wave32, WMMA bf16->f32)
// ---------------------------------------------------------------------------

typedef __bf16 bf16;
typedef __attribute__((ext_vector_type(16))) __bf16 v16bf;
typedef __attribute__((ext_vector_type(8)))  __bf16 v8bf;
typedef __attribute__((ext_vector_type(8)))  float  v8f;
typedef __attribute__((ext_vector_type(4)))  int    v4i;

#define DIMC   512
#define HIDC   2048
#define QKD    128
#define GRP    256
#define KSZ    17
#define PADT   8
#define BATCH  4
#define SEQ    4096
#define NTOK   (BATCH * SEQ)   // 16384

// ---------------- CDNA5 async global->LDS staging ----------------------------

#if __has_builtin(__builtin_amdgcn_global_load_async_to_lds_b128) && \
    __has_builtin(__builtin_amdgcn_s_wait_asynccnt)
#define HAVE_ASYNC_LDS 1
#else
#define HAVE_ASYNC_LDS 0
#endif

typedef __attribute__((address_space(1))) v4i* as1_v4i_p;   // global
typedef __attribute__((address_space(3))) v4i* as3_v4i_p;   // LDS

// copy 16 bytes (8 bf16) global -> LDS via GLOBAL_LOAD_ASYNC_TO_LDS_B128
__device__ __forceinline__ void stage16(const bf16* g, bf16* l) {
#if HAVE_ASYNC_LDS
    // builtin signature: (v4i AS1*, v4i AS3*, imm offset, imm cpol)
    // generic->AS pointer conversion via integer cast; generic LDS addresses
    // carry the LDS byte offset in the low 32 bits on AMDGPU.
    __builtin_amdgcn_global_load_async_to_lds_b128(
        (as1_v4i_p)(uintptr_t)g, (as3_v4i_p)(uintptr_t)l, 0, 0);
#else
    *(v8bf*)l = *(const v8bf*)g;
#endif
}
// wait for this wave's outstanding async LDS copies
__device__ __forceinline__ void stage_fence() {
#if HAVE_ASYNC_LDS
    __builtin_amdgcn_s_wait_asynccnt(0);
#endif
}

// ---------------- fast math --------------------------------------------------

__device__ __forceinline__ float fast_rcp(float x) {
#if __has_builtin(__builtin_amdgcn_rcpf)
    return __builtin_amdgcn_rcpf(x);
#else
    return 1.0f / x;
#endif
}
__device__ __forceinline__ float fast_sigmoid(float x) {
    return fast_rcp(1.0f + __expf(-x));
}
__device__ __forceinline__ float silu(float x) { return x * fast_sigmoid(x); }

// ---------------- WMMA helpers ----------------------------------------------

__device__ inline v8f zero8() {
    v8f z;
#pragma unroll
    for (int i = 0; i < 8; ++i) z[i] = 0.0f;
    return z;
}

// A-matrix 16x32 bf16 fragment: lane l holds row (l&15); K halves live at
// [8h, 8h+8) and [16+8h, 24+8h) where h = l>>4  (ISA 7.12.2, 16-bit A layout)
__device__ inline v16bf load_a_frag(const bf16* row, int hi) {
    v8bf lo = *(const v8bf*)(row + 8 * hi);
    v8bf hc = *(const v8bf*)(row + 16 + 8 * hi);
    v16bf r;
#pragma unroll
    for (int i = 0; i < 8; ++i) { r[i] = lo[i]; r[i + 8] = hc[i]; }
    return r;
}

// B-matrix 32x16 bf16 fragment from "Bt" (N x K row-major): lane l holds
// column (l&15); K halves live at [8h, 8h+8) and [16+8h, 24+8h)
__device__ inline v16bf load_b_frag(const bf16* row, int hi) {
    v8bf lo = *(const v8bf*)(row + 8 * hi);
    v8bf hc = *(const v8bf*)(row + 16 + 8 * hi);
    v16bf r;
#pragma unroll
    for (int i = 0; i < 8; ++i) { r[i] = lo[i]; r[i + 8] = hc[i]; }
    return r;
}

__device__ inline v8f wmma_bf16(v16bf a, v16bf b, v8f c) {
    return __builtin_amdgcn_wmma_f32_16x16x32_bf16(
        /*neg_a=*/false, a, /*neg_b=*/false, b,
        /*c_mod=*/(short)0, c, /*reuse_a=*/false, /*reuse_b=*/false);
}

// ---------------- 1) token shift + layernorm (both branches) ----------------

__global__ __launch_bounds__(256)
void prep_shift_ln_kernel(const float* __restrict__ x,
                          const float* __restrict__ hg, const float* __restrict__ hb,
                          const float* __restrict__ qg, const float* __restrict__ qb,
                          bf16* __restrict__ lnh, bf16* __restrict__ lnq) {
    const int t    = blockIdx.x;          // global token
    const int time = t & (SEQ - 1);
    const int tid  = threadIdx.x;         // 256 threads * 2 channels
    __shared__ float red[256];
    __shared__ float s_mu, s_rstd;

    float vals[2];
#pragma unroll
    for (int i = 0; i < 2; ++i) {
        const int c = tid * 2 + i;
        float v;
        if (c < DIMC / 2)   // shifted half
            v = (time > 0) ? x[(size_t)(t - 1) * DIMC + c] : 0.0f;
        else
            v = x[(size_t)t * DIMC + c];
        vals[i] = v;
    }
    red[tid] = vals[0] + vals[1];
    __syncthreads();
    for (int o = 128; o > 0; o >>= 1) {
        if (tid < o) red[tid] += red[tid + o];
        __syncthreads();
    }
    if (tid == 0) s_mu = red[0] * (1.0f / DIMC);
    __syncthreads();
    const float mu = s_mu;
    const float d0 = vals[0] - mu, d1 = vals[1] - mu;
    red[tid] = d0 * d0 + d1 * d1;
    __syncthreads();
    for (int o = 128; o > 0; o >>= 1) {
        if (tid < o) red[tid] += red[tid + o];
        __syncthreads();
    }
    if (tid == 0) s_rstd = rsqrtf(red[0] * (1.0f / DIMC) + 1e-5f);
    __syncthreads();
    const float rs = s_rstd;
#pragma unroll
    for (int i = 0; i < 2; ++i) {
        const int c = tid * 2 + i;
        const float nv = (vals[i] - mu) * rs;
        lnh[(size_t)t * DIMC + c] = (bf16)(nv * hg[c] + hb[c]);
        lnq[(size_t)t * DIMC + c] = (bf16)(nv * qg[c] + qb[c]);
    }
}

// ---------------- 2) generic fp32(RxC) -> bf16(CxR) transpose ----------------

__global__ __launch_bounds__(256)
void transpose_f32_bf16_kernel(const float* __restrict__ src, bf16* __restrict__ dst,
                               int R, int C) {
    __shared__ float tile[32][33];
    const int c0 = blockIdx.x * 32, r0 = blockIdx.y * 32;
    const int tx = threadIdx.x & 31, ty = threadIdx.x >> 5;   // ty: 0..7
#pragma unroll
    for (int i = 0; i < 32; i += 8) {
        const int r = r0 + ty + i, c = c0 + tx;
        tile[ty + i][tx] = (r < R && c < C) ? src[(size_t)r * C + c] : 0.0f;
    }
    __syncthreads();
#pragma unroll
    for (int i = 0; i < 32; i += 8) {
        const int r = c0 + ty + i, c = r0 + tx;   // dst is C x R
        if (r < C && c < R) dst[(size_t)r * R + c] = (bf16)tile[tx][ty + i];
    }
}

// ---------------- 3) WMMA GEMM: C = act(A(MxK) * Bt(NxK)^T + bias) -----------
// block: 256 thr (8 waves), 128x128 tile; wave tile 32x64 (2x4 accums).
// A/B tiles staged with GLOBAL_LOAD_ASYNC_TO_LDS_B128 when available.

template <int ACT>
__global__ __launch_bounds__(256)
void gemm_bias_act_kernel(const bf16* __restrict__ A, int lda,
                          const bf16* __restrict__ Bt, int ldb,
                          const float* __restrict__ bias,
                          bf16* __restrict__ C, int ldc, int K) {
    __shared__ bf16 As[128 * 32];
    __shared__ bf16 Bs[128 * 32];
    const int tid  = threadIdx.x;
    const int wave = tid >> 5, lane = tid & 31;
    const int hi = lane >> 4, r = lane & 15;
    const int m_blk = blockIdx.y * 128;
    const int n_blk = blockIdx.x * 128;
    const int wm = (wave >> 1) * 32;   // 0,32,64,96
    const int wn = (wave & 1) * 64;    // 0,64

    v8f acc[2][4];
#pragma unroll
    for (int mi = 0; mi < 2; ++mi)
#pragma unroll
        for (int ni = 0; ni < 4; ++ni) acc[mi][ni] = zero8();

    const int row_c = tid >> 1;          // 0..127
    const int hoff  = (tid & 1) * 16;    // 0 / 16
    const bf16* gA0 = A  + (size_t)(m_blk + row_c) * lda + hoff;
    const bf16* gB0 = Bt + (size_t)(n_blk + row_c) * ldb + hoff;
    bf16* lA = &As[row_c * 32 + hoff];
    bf16* lB = &Bs[row_c * 32 + hoff];

    for (int k0 = 0; k0 < K; k0 += 32) {
        stage16(gA0 + k0,     lA);
        stage16(gA0 + k0 + 8, lA + 8);
        stage16(gB0 + k0,     lB);
        stage16(gB0 + k0 + 8, lB + 8);
        if (k0 + 32 < K) {   // warm L2 for the next K tile (global_prefetch_b8)
            __builtin_prefetch(gA0 + k0 + 32, 0, 1);
            __builtin_prefetch(gB0 + k0 + 32, 0, 1);
        }
        stage_fence();
        __syncthreads();

        v16bf afrag[2], bfrag[4];
#pragma unroll
        for (int mi = 0; mi < 2; ++mi)
            afrag[mi] = load_a_frag(&As[(wm + mi * 16 + r) * 32], hi);
#pragma unroll
        for (int ni = 0; ni < 4; ++ni)
            bfrag[ni] = load_b_frag(&Bs[(wn + ni * 16 + r) * 32], hi);
#pragma unroll
        for (int mi = 0; mi < 2; ++mi)
#pragma unroll
            for (int ni = 0; ni < 4; ++ni)
                acc[mi][ni] = wmma_bf16(afrag[mi], bfrag[ni], acc[mi][ni]);
        __syncthreads();
    }

#pragma unroll
    for (int mi = 0; mi < 2; ++mi) {
#pragma unroll
        for (int ni = 0; ni < 4; ++ni) {
            const int n  = n_blk + wn + ni * 16 + r;
            const float bv = bias[n];
#pragma unroll
            for (int j = 0; j < 8; ++j) {
                const int m = m_blk + wm + mi * 16 + j + 8 * hi;
                float v = acc[mi][ni][j] + bv;
                if (ACT == 1) v = silu(v);
                C[(size_t)m * ldc + n] = (bf16)v;
            }
        }
    }
}

// ---------------- 4) depthwise conv (residual); FINAL adds x, fp32 out ------

template <int FINAL>
__global__ __launch_bounds__(256)
void dwconv_kernel(const bf16* __restrict__ y, const float* __restrict__ w, int C,
                   bf16* __restrict__ outb,
                   const float* __restrict__ xres, float* __restrict__ outf) {
    const size_t idx = (size_t)blockIdx.x * 256 + threadIdx.x;   // over NTOK*C
    const int    c   = (int)(idx % C);
    const long   t   = (long)(idx / C);
    const int    tm  = (int)(t & (SEQ - 1));
    float acc = 0.0f;
#pragma unroll
    for (int j = 0; j < KSZ; ++j) {
        const int tt = tm + j - PADT;
        if (tt >= 0 && tt < SEQ)
            acc += (float)y[(size_t)(t + j - PADT) * C + c] * w[c * KSZ + j];
    }
    const float res = (float)y[(size_t)t * C + c] + acc;
    if (FINAL)
        outf[(size_t)t * C + c] = xres[(size_t)t * C + c] + res;
    else
        outb[(size_t)t * C + c] = (bf16)res;
}

// ---------------- 5) qk -> 4 scaled variants (quad_q, lin_q, quad_k, lin_k) -

__global__ __launch_bounds__(256)
void qscale_kernel(const bf16* __restrict__ qk, const float* __restrict__ gamma,
                   const float* __restrict__ beta, bf16* __restrict__ out4) {
    const size_t idx = (size_t)blockIdx.x * 256 + threadIdx.x;  // 4*NTOK*QKD
    const int    d   = (int)(idx % QKD);
    const size_t tv  = idx / QKD;
    const int    var = (int)(tv / NTOK);
    const size_t t   = tv % NTOK;
    out4[idx] = (bf16)((float)qk[t * QKD + d] * gamma[var * QKD + d] +
                       beta[var * QKD + d]);
}

// ---------------- 6) lin_kv / lin_ku = (1/n) * lk^T @ {v,u} ------------------
// block = (batch, 64-wide e tile); M=128 (d), K=SEQ; both operands transposed
// through LDS each 32-token step.

__global__ __launch_bounds__(256)
void linkv_kernel(const bf16* __restrict__ lk,       // [NTOK][QKD]
                  const bf16* __restrict__ h_post,   // [NTOK][HID]: v|u halves
                  float* __restrict__ lin_kv, float* __restrict__ lin_ku) {
    const int batch = blockIdx.x >> 4;
    const int et    = blockIdx.x & 15;
    const int e0    = et * 64;
    __shared__ bf16 LkT[QKD * 32];   // [d][t]
    __shared__ bf16 VT[64 * 32];     // [e][t]
    __shared__ bf16 UT[64 * 32];
    const int tid = threadIdx.x, wave = tid >> 5, lane = tid & 31;
    const int hi = lane >> 4, r = lane & 15;
    const size_t tok0 = (size_t)batch * SEQ;

    v8f accv[4], accu[4];
#pragma unroll
    for (int ni = 0; ni < 4; ++ni) { accv[ni] = zero8(); accu[ni] = zero8(); }

    for (int k0 = 0; k0 < SEQ; k0 += 32) {
        {   // stage lk^T : 32 tokens x 128 d
            const int tt = tid >> 3, d0 = (tid & 7) * 16;
            const bf16* src = lk + (tok0 + k0 + tt) * QKD + d0;
#pragma unroll
            for (int i = 0; i < 16; ++i) LkT[(d0 + i) * 32 + tt] = src[i];
        }
        {   // stage v^T, u^T : 32 tokens x 64 e
            const int tt = tid >> 3, ee = (tid & 7) * 8;
            const bf16* sv = h_post + (tok0 + k0 + tt) * HIDC + e0 + ee;
            const bf16* su = sv + 1024;
#pragma unroll
            for (int i = 0; i < 8; ++i) {
                VT[(ee + i) * 32 + tt] = sv[i];
                UT[(ee + i) * 32 + tt] = su[i];
            }
        }
        __syncthreads();
        v16bf a = load_a_frag(&LkT[(wave * 16 + r) * 32], hi);
#pragma unroll
        for (int ni = 0; ni < 4; ++ni) {
            v16bf bv = load_b_frag(&VT[(ni * 16 + r) * 32], hi);
            accv[ni] = wmma_bf16(a, bv, accv[ni]);
            v16bf bu = load_b_frag(&UT[(ni * 16 + r) * 32], hi);
            accu[ni] = wmma_bf16(a, bu, accu[ni]);
        }
        __syncthreads();
    }
    const float inv_n = 1.0f / (float)SEQ;
#pragma unroll
    for (int ni = 0; ni < 4; ++ni) {
        const int e = e0 + ni * 16 + r;
#pragma unroll
        for (int j = 0; j < 8; ++j) {
            const int d = wave * 16 + j + 8 * hi;
            const size_t o = ((size_t)batch * 128 + d) * 1024 + e;
            lin_kv[o] = accv[ni][j] * inv_n;
            lin_ku[o] = accu[ni][j] * inv_n;
        }
    }
}

// ---------------- 7) fused quadratic + linear attention ----------------------
// block = (group gi, 32-row block rb). sim=relu(qq k^T/G)^2 -> bf16 in LDS,
// then per 64-wide e tile: att = attn @ v^T (K=256) + lq @ lin_kvT^T (K=128).

__global__ __launch_bounds__(256)
void attention_kernel(const bf16* __restrict__ qkt4,
                      const bf16* __restrict__ h_post,
                      const bf16* __restrict__ linkvT,
                      const bf16* __restrict__ linkuT,
                      bf16* __restrict__ att_v, bf16* __restrict__ att_u) {
    const int gi = blockIdx.x >> 3;        // 0..63
    const int rb = blockIdx.x & 7;
    const int batch = gi >> 4;
    const size_t gtok0 = (size_t)gi * GRP;
    const size_t rtok0 = gtok0 + (size_t)rb * 32;

    __shared__ bf16 Qs[32 * 128];     // qq rows, later lq rows
    __shared__ bf16 attnS[32 * 256];  // relu(sim/G)^2 in bf16
    __shared__ bf16 VT[64 * 256];     // v/u tile transposed: [e][j]

    const bf16* quad_q = qkt4;
    const bf16* lin_q  = qkt4 + (size_t)1 * NTOK * QKD;
    const bf16* quad_k = qkt4 + (size_t)2 * NTOK * QKD;

    const int tid = threadIdx.x, wave = tid >> 5, lane = tid & 31;
    const int hi = lane >> 4, r = lane & 15;
    const int srow = tid >> 3, soff = (tid & 7) * 16;

    {   // stage qq rows: 32 x 128 (async direct-to-LDS when available)
        const bf16* src = quad_q + (rtok0 + srow) * QKD + soff;
        stage16(src,     &Qs[srow * 128 + soff]);
        stage16(src + 8, &Qs[srow * 128 + soff + 8]);
        stage_fence();
    }
    __syncthreads();

    {   // sim phase: 32 x 256; wave: m strip (wave>>2), 4 n strips
        const int wm = (wave >> 2) * 16;
        v8f acc[4];
#pragma unroll
        for (int q = 0; q < 4; ++q) acc[q] = zero8();
        for (int k0 = 0; k0 < QKD; k0 += 32) {
            v16bf a = load_a_frag(&Qs[(wm + r) * 128 + k0], hi);
#pragma unroll
            for (int q = 0; q < 4; ++q) {
                const int jrow = ((wave & 3) * 4 + q) * 16 + r;      // col j
                const bf16* brow = quad_k + (gtok0 + jrow) * QKD + k0;
                v16bf b = load_b_frag(brow, hi);
                acc[q] = wmma_bf16(a, b, acc[q]);
            }
        }
#pragma unroll
        for (int q = 0; q < 4; ++q) {
            const int jb = ((wave & 3) * 4 + q) * 16;
#pragma unroll
            for (int j = 0; j < 8; ++j) {
                const int m = wm + j + 8 * hi;
                const int n = jb + r;
                float s = acc[q][j] * (1.0f / GRP);
                s = fmaxf(s, 0.0f);
                attnS[m * 256 + n] = (bf16)(s * s);
            }
        }
    }
    __syncthreads();

    {   // restage Qs with lq rows
        const bf16* src = lin_q + (rtok0 + srow) * QKD + soff;
        stage16(src,     &Qs[srow * 128 + soff]);
        stage16(src + 8, &Qs[srow * 128 + soff + 8]);
        stage_fence();
    }
    __syncthreads();

    for (int et = 0; et < 16; ++et) {
        const int e0 = et * 64;
        for (int pass = 0; pass < 2; ++pass) {     // 0: v, 1: u
            {   // stage VT[e][j] = h_post[token j][pass*1024 + e0 + e]
                const int j = tid;                 // 0..255
                const bf16* src = h_post + (gtok0 + j) * HIDC + pass * 1024 + e0;
#pragma unroll
                for (int blk = 0; blk < 8; ++blk) {
                    v8bf ch = *(const v8bf*)(src + blk * 8);
#pragma unroll
                    for (int i = 0; i < 8; ++i)
                        VT[(blk * 8 + i) * 256 + j] = ch[i];
                }
            }
            __syncthreads();

            const int wm = (wave >> 2) * 16;       // 0/16
            const int wn = (wave & 3) * 16;        // 0..48
            v8f acc = zero8();
            // quadratic part: K = 256
            for (int k0 = 0; k0 < GRP; k0 += 32) {
                v16bf a = load_a_frag(&attnS[(wm + r) * 256 + k0], hi);
                v16bf b = load_b_frag(&VT[(wn + r) * 256 + k0], hi);
                acc = wmma_bf16(a, b, acc);
            }
            // linear part: K = 128
            const bf16* lkt = (pass == 0) ? linkvT : linkuT;
            for (int k0 = 0; k0 < QKD; k0 += 32) {
                v16bf a = load_a_frag(&Qs[(wm + r) * 128 + k0], hi);
                const bf16* brow =
                    lkt + ((size_t)batch * 1024 + e0 + wn + r) * QKD + k0;
                v16bf b = load_b_frag(brow, hi);
                acc = wmma_bf16(a, b, acc);
            }
            bf16* outp = (pass == 0) ? att_v : att_u;
            const int n = e0 + wn + r;
#pragma unroll
            for (int j = 0; j < 8; ++j) {
                const int m = wm + j + 8 * hi;
                outp[(rtok0 + m) * 1024 + n] = (bf16)acc[j];
            }
            __syncthreads();
        }
    }
}

// ---------------- 8) gated combination ---------------------------------------

__global__ __launch_bounds__(256)
void gate_kernel(const bf16* __restrict__ att_v, const bf16* __restrict__ att_u,
                 const bf16* __restrict__ h_post, bf16* __restrict__ out) {
    const size_t idx = (size_t)blockIdx.x * 256 + threadIdx.x;  // NTOK*1024
    const int    c   = (int)(idx % 1024);
    const size_t t   = idx / 1024;
    const float v  = (float)h_post[t * HIDC + c];
    const float u  = (float)h_post[t * HIDC + 1024 + c];
    const float av = (float)att_v[idx];
    const float au = (float)att_u[idx];
    out[idx] = (bf16)(au * v * fast_sigmoid(av * u));
}

// ---------------- 9) generic layernorm (bf16 in/out) -------------------------

__global__ __launch_bounds__(256)
void ln_kernel(const bf16* __restrict__ in, const float* __restrict__ g,
               const float* __restrict__ b, bf16* __restrict__ out, int C) {
    const int t = blockIdx.x, tid = threadIdx.x;
    const int cpt = C >> 8;                 // channels per thread (<=4)
    __shared__ float red[256];
    __shared__ float s_mu, s_rstd;
    float vals[4];
    float s = 0.0f;
    for (int i = 0; i < cpt; ++i) {
        const int c = tid * cpt + i;
        vals[i] = (float)in[(size_t)t * C + c];
        s += vals[i];
    }
    red[tid] = s;
    __syncthreads();
    for (int o = 128; o > 0; o >>= 1) {
        if (tid < o) red[tid] += red[tid + o];
        __syncthreads();
    }
    if (tid == 0) s_mu = red[0] / (float)C;
    __syncthreads();
    const float mu = s_mu;
    s = 0.0f;
    for (int i = 0; i < cpt; ++i) { const float d = vals[i] - mu; s += d * d; }
    red[tid] = s;
    __syncthreads();
    for (int o = 128; o > 0; o >>= 1) {
        if (tid < o) red[tid] += red[tid + o];
        __syncthreads();
    }
    if (tid == 0) s_rstd = rsqrtf(red[0] / (float)C + 1e-5f);
    __syncthreads();
    const float rs = s_rstd;
    for (int i = 0; i < cpt; ++i) {
        const int c = tid * cpt + i;
        out[(size_t)t * C + c] = (bf16)((vals[i] - mu) * rs * g[c] + b[c]);
    }
}

// ---------------------------------------------------------------------------
// host launcher
// ---------------------------------------------------------------------------

extern "C" void kernel_launch(void* const* d_in, const int* in_sizes, int n_in,
                              void* d_out, int out_size, void* d_ws, size_t ws_size,
                              hipStream_t stream) {
    (void)in_sizes; (void)n_in; (void)out_size; (void)ws_size;
    const float* x      = (const float*)d_in[0];
    const float* h_ln_g = (const float*)d_in[1];
    const float* h_ln_b = (const float*)d_in[2];
    const float* h_W    = (const float*)d_in[3];
    const float* h_b    = (const float*)d_in[4];
    const float* h_conv = (const float*)d_in[5];
    const float* qk_ln_g= (const float*)d_in[6];
    const float* qk_ln_b= (const float*)d_in[7];
    const float* qk_W   = (const float*)d_in[8];
    const float* qk_b   = (const float*)d_in[9];
    const float* qk_conv= (const float*)d_in[10];
    const float* os_gam = (const float*)d_in[11];
    const float* os_bet = (const float*)d_in[12];
    const float* o_ln_g = (const float*)d_in[13];
    const float* o_ln_b = (const float*)d_in[14];
    const float* o_W    = (const float*)d_in[15];
    const float* o_b    = (const float*)d_in[16];
    const float* o_conv = (const float*)d_in[17];
    float* out = (float*)d_out;

    // bump allocator over workspace
    char* wsb = (char*)d_ws;
    size_t off = 0;
    auto alloc = [&](size_t bytes) -> char* {
        char* p = wsb + off;
        off = (off + bytes + 255) & ~(size_t)255;
        return p;
    };
    bf16*  lnh     = (bf16*)alloc((size_t)NTOK * DIMC * 2);
    bf16*  lnq     = (bf16*)alloc((size_t)NTOK * DIMC * 2);
    bf16*  WtH     = (bf16*)alloc((size_t)HIDC * DIMC * 2);
    bf16*  WtQ     = (bf16*)alloc((size_t)QKD  * DIMC * 2);
    bf16*  WtO     = (bf16*)alloc((size_t)DIMC * 1024 * 2);
    bf16*  h_pre   = (bf16*)alloc((size_t)NTOK * HIDC * 2);
    bf16*  h_post  = (bf16*)alloc((size_t)NTOK * HIDC * 2);
    bf16*  qk_pre  = (bf16*)alloc((size_t)NTOK * QKD * 2);
    bf16*  qk_post = (bf16*)alloc((size_t)NTOK * QKD * 2);
    bf16*  qkt4    = (bf16*)alloc((size_t)4 * NTOK * QKD * 2);
    float* linkv   = (float*)alloc((size_t)BATCH * 128 * 1024 * 4);
    float* linku   = (float*)alloc((size_t)BATCH * 128 * 1024 * 4);
    bf16*  linkvT  = (bf16*)alloc((size_t)BATCH * 1024 * 128 * 2);
    bf16*  linkuT  = (bf16*)alloc((size_t)BATCH * 1024 * 128 * 2);
    bf16*  attv    = (bf16*)alloc((size_t)NTOK * 1024 * 2);
    bf16*  attu    = (bf16*)alloc((size_t)NTOK * 1024 * 2);
    bf16*  gated   = (bf16*)alloc((size_t)NTOK * 1024 * 2);
    bf16*  lno     = (bf16*)alloc((size_t)NTOK * 1024 * 2);
    bf16*  yo      = (bf16*)alloc((size_t)NTOK * DIMC * 2);

    // 1) token shift + LN (both branches)
    prep_shift_ln_kernel<<<NTOK, 256, 0, stream>>>(x, h_ln_g, h_ln_b,
                                                   qk_ln_g, qk_ln_b, lnh, lnq);
    // 2) weight transposes (fp32 KxN -> bf16 NxK)
    transpose_f32_bf16_kernel<<<dim3(HIDC / 32, DIMC / 32), 256, 0, stream>>>(
        h_W, WtH, DIMC, HIDC);
    transpose_f32_bf16_kernel<<<dim3(QKD / 32, DIMC / 32), 256, 0, stream>>>(
        qk_W, WtQ, DIMC, QKD);
    transpose_f32_bf16_kernel<<<dim3(DIMC / 32, 1024 / 32), 256, 0, stream>>>(
        o_W, WtO, 1024, DIMC);
    // 3) h = silu(ln_h @ h_W + h_b)
    gemm_bias_act_kernel<1><<<dim3(HIDC / 128, NTOK / 128), 256, 0, stream>>>(
        lnh, DIMC, WtH, DIMC, h_b, h_pre, HIDC, DIMC);
    // 4) h_post = h + dwconv(h)
    dwconv_kernel<0><<<(NTOK * (size_t)HIDC) / 256, 256, 0, stream>>>(
        h_pre, h_conv, HIDC, h_post, nullptr, nullptr);
    // 5) qk branch
    gemm_bias_act_kernel<1><<<dim3(QKD / 128, NTOK / 128), 256, 0, stream>>>(
        lnq, DIMC, WtQ, DIMC, qk_b, qk_pre, QKD, DIMC);
    dwconv_kernel<0><<<(NTOK * (size_t)QKD) / 256, 256, 0, stream>>>(
        qk_pre, qk_conv, QKD, qk_post, nullptr, nullptr);
    // 6) scaled variants
    qscale_kernel<<<((size_t)4 * NTOK * QKD) / 256, 256, 0, stream>>>(
        qk_post, os_gam, os_bet, qkt4);
    // 7) linear-attention kv summaries (uses lin_k = variant 3)
    linkv_kernel<<<BATCH * 16, 256, 0, stream>>>(
        qkt4 + (size_t)3 * NTOK * QKD, h_post, linkv, linku);
    // 8) transpose kv summaries per batch (fp32 128x1024 -> bf16 1024x128)
    for (int b = 0; b < BATCH; ++b) {
        transpose_f32_bf16_kernel<<<dim3(1024 / 32, 128 / 32), 256, 0, stream>>>(
            linkv + (size_t)b * 128 * 1024, linkvT + (size_t)b * 1024 * 128,
            128, 1024);
        transpose_f32_bf16_kernel<<<dim3(1024 / 32, 128 / 32), 256, 0, stream>>>(
            linku + (size_t)b * 128 * 1024, linkuT + (size_t)b * 1024 * 128,
            128, 1024);
    }
    // 9) fused quadratic + linear attention
    attention_kernel<<<64 * 8, 256, 0, stream>>>(qkt4, h_post, linkvT, linkuT,
                                                 attv, attu);
    // 10) shared-A gated combination
    gate_kernel<<<((size_t)NTOK * 1024) / 256, 256, 0, stream>>>(
        attv, attu, h_post, gated);
    // 11) output FFConvM: LN -> GEMM+SiLU -> conv + residual(+x)
    ln_kernel<<<NTOK, 256, 0, stream>>>(gated, o_ln_g, o_ln_b, lno, 1024);
    gemm_bias_act_kernel<1><<<dim3(DIMC / 128, NTOK / 128), 256, 0, stream>>>(
        lno, 1024, WtO, 1024, o_b, yo, DIMC, 1024);
    dwconv_kernel<1><<<((size_t)NTOK * DIMC) / 256, 256, 0, stream>>>(
        yo, o_conv, DIMC, nullptr, x, out);
}
